// EfficientParticleNet_21973052686571
// MI455X (gfx1250) — compile-verified, matrix operand order checked by reading, not run on hardware
//
#include <hip/hip_runtime.h>

#define EPS_BN 1e-5f
#define EPS_GN 1e-5f

typedef __attribute__((ext_vector_type(16))) _Float16 v16h;
typedef __attribute__((ext_vector_type(8)))  float    v8f;

#define WMMA(a,b,c) __builtin_amdgcn_wmma_f32_16x16x32_f16(false,(a),false,(b),(short)0,(c),false,false)

__device__ __forceinline__ float lrelu(float x) { return x > 0.f ? x : 0.01f * x; }

// ---- WMMA fragment loader -------------------------------------------------
// 16-bit A-matrix 16x32 layout (ISA 7.12.2): lane L -> row (L&15), half h=L>>4.
// VGPR v, pos p: K = (v>=4?16:0) + 8*h + 2*(v&3) + p.
// B (32x16) mirrors this with lane&15 = column index, so the same loader
// serves A tiles (row-major activations) and B tiles (row-major W^T).
// Each 8-half group is contiguous -> compiler merges into ds_load_b128.
__device__ __forceinline__ v16h load_frag(const _Float16* __restrict__ p0,
                                          int stride, int row0, int kbase, int lane) {
  const int r = row0 + (lane & 15);
  const int h = (lane >> 4) & 1;
  const _Float16* p = p0 + r * stride + kbase + 8 * h;
  v16h f;
#pragma unroll
  for (int v = 0; v < 8; ++v) {
    const int k = ((v & 4) ? 16 : 0) + ((v & 3) << 1);
    f[2 * v]     = p[k];
    f[2 * v + 1] = p[k + 1];
  }
  return f;
}

// ---- weight prep: fp32 row-major (din x dout) -> fp16 transposed (dout x din)
__global__ __launch_bounds__(256)
void prep_wt(const float* __restrict__ W, const float* __restrict__ Wsub,
             _Float16* __restrict__ dst, int din, int dout) {
  int i = blockIdx.x * 256 + threadIdx.x;
  if (i >= din * dout) return;
  int o = i / din, k = i - o * din;
  float v = W[k * dout + o];
  if (Wsub) v -= Wsub[k * dout + o];
  dst[o * din + k] = (_Float16)v;
}

// ---- GraphNorm ------------------------------------------------------------
__global__ __launch_bounds__(64)
void graphnorm_kernel(const float* __restrict__ x, const float* __restrict__ w,
                      const float* __restrict__ bb, const float* __restrict__ ms,
                      float* __restrict__ h0) {
  __shared__ float xs[64][10];
  __shared__ float mn[9], inv[9];
  int b = blockIdx.x, n = threadIdx.x;
  for (int f = 0; f < 9; ++f) xs[n][f] = x[((size_t)b * 64 + n) * 9 + f];
  __syncthreads();
  if (n < 9) {
    float m = 0.f;
    for (int i = 0; i < 64; ++i) m += xs[i][n];
    m *= (1.f / 64.f);
    float msm = ms[n] * m;
    float var = 0.f;
    for (int i = 0; i < 64; ++i) { float d = xs[i][n] - msm; var += d * d; }
    var *= (1.f / 64.f);
    mn[n] = msm;
    inv[n] = w[n] * rsqrtf(var + EPS_GN);
  }
  __syncthreads();
  for (int f = 0; f < 9; ++f)
    h0[((size_t)b * 64 + n) * 9 + f] = (xs[n][f] - mn[f]) * inv[f] + bb[f];
}

// ---- EdgeConv kernels -----------------------------------------------------
struct ConvArgs {
  const float *b1, *b2, *b3, *bsc;
  const float *bn1g, *bn1b, *bn1m, *bn1v;
  const float *bn2g, *bn2b, *bn2m, *bn2v;
  const float *bn3g, *bn3b, *bn3m, *bn3v;
  const float *scg, *scb, *scm, *scv;
  const _Float16 *W1dT, *W1bT, *W2T, *W3T, *WscT;
  const float *W1raw, *Wscraw;  // conv1 VALU path (cin=9)
};

// Padded LDS strides to break 64-bank alignment (192 words == 0 mod 64).
#define SF32 202   // f32 node buffers: 202 mod 64 = 10 -> conflict-light
#define SF16 200   // f16 buffers: 400B rows keep 16B align for b128 merges
#define SD2  66

// LDS layout (bytes)
#define LDS_ST   0        // 8 x 192 f32 BN scale/shift tables          (6144)
#define LDS_OUT  6144     // 64xSF32 f32 node out (init = shortcut)     (51712)
#define LDS_A    57856    // 64xSF32 f32 per-node term A                (51712)
#define LDS_BP   109568   // 64xSF32 f32 per-node term Bp               (51712)
#define LDS_E2   57856    // 256xSF16 f16, overlays A+Bp                (102400)
#define LDS_E1   161280   // 256xSF16 f16 edge activations              (102400)
#define LDS_X    263680   // dyn: 64xSF16 f16 X | first: 64x12 f32 x    (25600/3072)
#define LDS_SQ   289280   // 64 f32                                     (256)
#define LDS_D2   289536   // 64xSD2 f32 pairwise dist                   (16896)
#define LDS_IDX  306432   // 64x4 int                                   (1024)
#define SMEM_DYN   307456
#define SMEM_FIRST 266752

template <bool FIRST>
__global__ __launch_bounds__(256)
void conv_kernel(const float* __restrict__ hin, float* __restrict__ hout, ConvArgs cw) {
  extern __shared__ char smem[];
  float* st  = (float*)(smem + LDS_ST);
  float* s1 = st;       float* t1 = st + 192;
  float* s2 = st + 384; float* t2 = st + 576;
  float* s3 = st + 768; float* t3 = st + 960;
  float* ssc = st + 1152; float* tsc = st + 1344;
  float* outb = (float*)(smem + LDS_OUT);
  float* A    = (float*)(smem + LDS_A);
  float* Bp   = (float*)(smem + LDS_BP);
  _Float16* E1 = (_Float16*)(smem + LDS_E1);
  _Float16* E2 = (_Float16*)(smem + LDS_E2);

  const int b = blockIdx.x;
  const int tid = threadIdx.x;
  const int lane = tid & 31;
  const int wave = tid >> 5;

  // Warm L2/L1 with the WMMA weight matrices (global_prefetch_b8 path).
#pragma unroll 1
  for (int off = tid * 128; off < 73728; off += 256 * 128) {
    __builtin_prefetch((const char*)cw.W2T + off, 0, 1);
    __builtin_prefetch((const char*)cw.W3T + off, 0, 1);
    if constexpr (!FIRST) {
      __builtin_prefetch((const char*)cw.W1dT + off, 0, 1);
      __builtin_prefetch((const char*)cw.W1bT + off, 0, 1);
      __builtin_prefetch((const char*)cw.WscT + off, 0, 1);
    }
  }

  // BN scale/shift tables (folded affine)
  for (int c = tid; c < 192; c += 256) {
    float s;
    s = cw.bn1g[c] * rsqrtf(cw.bn1v[c] + EPS_BN); s1[c] = s; t1[c] = cw.bn1b[c] - cw.bn1m[c] * s;
    s = cw.bn2g[c] * rsqrtf(cw.bn2v[c] + EPS_BN); s2[c] = s; t2[c] = cw.bn2b[c] - cw.bn2m[c] * s;
    s = cw.bn3g[c] * rsqrtf(cw.bn3v[c] + EPS_BN); s3[c] = s; t3[c] = cw.bn3b[c] - cw.bn3m[c] * s;
    s = cw.scg[c]  * rsqrtf(cw.scv[c]  + EPS_BN); ssc[c] = s; tsc[c] = cw.scb[c] - cw.scm[c] * s;
  }

  if constexpr (FIRST) {
    float* xg = (float*)(smem + LDS_X);
    for (int p = tid; p < 64 * 9; p += 256) {
      int i = p / 9, f = p - i * 9;
      xg[i * 12 + f] = hin[((size_t)b * 64 + i) * 9 + f];
    }
    __syncthreads();
    // node-level linear parts (cin=9: VALU) + shortcut
    for (int p = tid; p < 64 * 192; p += 256) {
      int i = p / 192, c = p - i * 192;
      const float* xr = xg + i * 12;
      float a = cw.b1[c], bp = 0.f, sc = cw.bsc[c];
#pragma unroll
      for (int f = 0; f < 9; ++f) {
        float xv = xr[f];
        float w1b = cw.W1raw[(f + 9) * 192 + c];
        a  += xv * (cw.W1raw[f * 192 + c] - w1b);
        bp += xv * w1b;
        sc += xv * cw.Wscraw[f * 192 + c];
      }
      A[i * SF32 + c] = a;
      Bp[i * SF32 + c] = bp;
      outb[i * SF32 + c] = sc * ssc[c] + tsc[c];
    }
    __syncthreads();
    // edge layer 1 (ring edges: node i <- j=(i+k+1)%64), row r = k*64+i
    {
      int r = tid, i = r & 63, k = r >> 6, j = (i + k + 1) & 63;
      const float* Ai = A + i * SF32;
      const float* Bj = Bp + j * SF32;
      _Float16* e = E1 + r * SF16;
      for (int c = 0; c < 192; ++c) {
        float v = lrelu(Ai[c] + Bj[c]);
        e[c] = (_Float16)(v * s1[c] + t1[c]);
      }
    }
  } else {
    _Float16* Xh = (_Float16*)(smem + LDS_X);
    float* sq = (float*)(smem + LDS_SQ);
    float* d2 = (float*)(smem + LDS_D2);
    int* idxL = (int*)(smem + LDS_IDX);
    const float* xin = hin + (size_t)b * 64 * 192;
    for (int p = tid; p < 64 * 192; p += 256) {
      int i = p / 192, c = p - i * 192;
      Xh[i * SF16 + c] = (_Float16)xin[p];
    }
    if (tid < 64) {
      const float* xr = xin + tid * 192;
      float s = 0.f;
      for (int c = 0; c < 192; ++c) s += xr[c] * xr[c];
      sq[tid] = s;
    }
    __syncthreads();
    // Gram = X X^T via WMMA (X as both A and B operands), then d2
    for (int t = wave; t < 16; t += 8) {
      int mt = t >> 2, nt = t & 3;
      v8f acc = {};
#pragma unroll
      for (int ks = 0; ks < 6; ++ks) {
        v16h af = load_frag(Xh, SF16, mt * 16, ks * 32, lane);
        v16h bf = load_frag(Xh, SF16, nt * 16, ks * 32, lane);
        acc = WMMA(af, bf, acc);
      }
      int n = lane & 15, h = (lane >> 4) & 1;
      int j = nt * 16 + n;
#pragma unroll
      for (int v = 0; v < 8; ++v) {
        int i = mt * 16 + v + 8 * h;
        float d = sq[i] + sq[j] - 2.f * acc[v];
        if (i == j) d += 1e10f;
        d2[i * SD2 + j] = d;
      }
    }
    __syncthreads();
    // top-K=4 smallest per row
    if (tid < 64) {
      float* row = d2 + tid * SD2;
      for (int k = 0; k < 4; ++k) {
        float best = 3.0e38f; int bj = 0;
        for (int j = 0; j < 64; ++j) { float v = row[j]; if (v < best) { best = v; bj = j; } }
        idxL[tid * 4 + k] = bj;
        row[bj] = 3.0e38f;
      }
    }
    __syncthreads();
    // node GEMMs: A = X@(W1a-W1b)+b1, Bp = X@W1b, sc = bn(X@Wsc+bsc)
    // grouped by (mode,nt) so the 6 weight B-fragments are reused over 4 M-tiles
    for (int g = wave; g < 36; g += 8) {
      int mode = g / 12, nt = g - mode * 12;
      const _Float16* WT = (mode == 0) ? cw.W1dT : (mode == 1) ? cw.W1bT : cw.WscT;
      v16h bf[6];
#pragma unroll
      for (int ks = 0; ks < 6; ++ks) bf[ks] = load_frag(WT, 192, nt * 16, ks * 32, lane);
      int n = lane & 15, h = (lane >> 4) & 1, col = nt * 16 + n;
      float* dst = (mode == 0) ? A : (mode == 1) ? Bp : outb;
      float add = (mode == 0) ? cw.b1[col] : (mode == 2) ? cw.bsc[col] : 0.f;
      for (int mt = 0; mt < 4; ++mt) {
        v8f acc = {};
#pragma unroll
        for (int ks = 0; ks < 6; ++ks) {
          v16h af = load_frag(Xh, SF16, mt * 16, ks * 32, lane);
          acc = WMMA(af, bf[ks], acc);
        }
#pragma unroll
        for (int v = 0; v < 8; ++v) {
          int i = mt * 16 + v + 8 * h;
          float y = acc[v] + add;
          if (mode == 2) y = y * ssc[col] + tsc[col];
          dst[i * SF32 + col] = y;
        }
      }
    }
    __syncthreads();
    // edge layer 1 (dynamic kNN edges)
    {
      int r = tid, i = r & 63, k = r >> 6, j = idxL[i * 4 + k];
      const float* Ai = A + i * SF32;
      const float* Bj = Bp + j * SF32;
      _Float16* e = E1 + r * SF16;
      for (int c = 0; c < 192; ++c) {
        float v = lrelu(Ai[c] + Bj[c]);
        e[c] = (_Float16)(v * s1[c] + t1[c]);
      }
    }
  }
  __syncthreads();
  // edge layer 2: E2 = bn2(lrelu(E1 @ W2 + b2)), 256x192 @ 192x192
  // nt outer: weight fragments register-resident across 2 M-tiles per wave
  for (int nt = 0; nt < 12; ++nt) {
    v16h bf[6];
#pragma unroll
    for (int ks = 0; ks < 6; ++ks) bf[ks] = load_frag(cw.W2T, 192, nt * 16, ks * 32, lane);
    int n = lane & 15, h = (lane >> 4) & 1, col = nt * 16 + n;
    float bb = cw.b2[col], s = s2[col], tt = t2[col];
    for (int mt = wave; mt < 16; mt += 8) {
      v8f acc = {};
#pragma unroll
      for (int ks = 0; ks < 6; ++ks) {
        v16h af = load_frag(E1, SF16, mt * 16, ks * 32, lane);
        acc = WMMA(af, bf[ks], acc);
      }
#pragma unroll
      for (int v = 0; v < 8; ++v) {
        int row = mt * 16 + v + 8 * h;
        float y = lrelu(acc[v] + bb);
        E2[row * SF16 + col] = (_Float16)(y * s + tt);
      }
    }
  }
  __syncthreads();
  // edge layer 3 + K-mean aggregation (bn3 affine commutes with mean)
  for (int g = wave; g < 48; g += 8) {
    int ti = g / 12, nt = g - ti * 12;
    v16h bf[6];
#pragma unroll
    for (int ks = 0; ks < 6; ++ks) bf[ks] = load_frag(cw.W3T, 192, nt * 16, ks * 32, lane);
    int n = lane & 15, h = (lane >> 4) & 1, col = nt * 16 + n;
    float bb = cw.b3[col];
    float sum[8] = {0.f, 0.f, 0.f, 0.f, 0.f, 0.f, 0.f, 0.f};
    for (int ke = 0; ke < 4; ++ke) {
      v8f acc = {};
#pragma unroll
      for (int ks = 0; ks < 6; ++ks) {
        v16h af = load_frag(E2, SF16, ke * 64 + ti * 16, ks * 32, lane);
        acc = WMMA(af, bf[ks], acc);
      }
#pragma unroll
      for (int v = 0; v < 8; ++v) sum[v] += lrelu(acc[v] + bb);
    }
    float s = s3[col], tt = t3[col];
#pragma unroll
    for (int v = 0; v < 8; ++v) {
      int i = ti * 16 + v + 8 * h;
      outb[i * SF32 + col] += (sum[v] * 0.25f) * s + tt;
    }
  }
  __syncthreads();
  {
    float* dst = hout + (size_t)b * 64 * 192;
    for (int p = tid; p < 64 * 192; p += 256) {
      int i = p / 192, c = p - i * 192;
      dst[p] = outb[i * SF32 + c];
    }
  }
}

// ---- attention pool + concat + bn0 ---------------------------------------
__global__ __launch_bounds__(64)
void attpool_kernel(const float* __restrict__ hin, const float* __restrict__ gin,
                    const float* Wa1, const float* ba1, const float* Wa2, const float* ba2,
                    const float* g0, const float* b0, const float* m0, const float* v0,
                    float* __restrict__ pooled) {
  __shared__ float aw[64];
  __shared__ float red;
  int b = blockIdx.x, n = threadIdx.x;
  const float* x = hin + ((size_t)b * 64 + n) * 192;
  float a = ba2[0];
  for (int c2 = 0; c2 < 48; ++c2) {
    float hsum = ba1[c2];
    for (int c = 0; c < 192; ++c) hsum += x[c] * Wa1[c * 48 + c2];
    a += lrelu(hsum) * Wa2[c2];
  }
  aw[n] = a;
  __syncthreads();
  if (n == 0) { float m = aw[0]; for (int i = 1; i < 64; ++i) m = fmaxf(m, aw[i]); red = m; }
  __syncthreads();
  float e = expf(a - red);
  aw[n] = e;
  __syncthreads();
  if (n == 0) { float s = 0.f; for (int i = 0; i < 64; ++i) s += aw[i]; red = s; }
  __syncthreads();
  float w = e / red;
  aw[n] = w;
  __syncthreads();
  for (int c = n; c < 196; c += 64) {
    float z;
    if (c < 192) {
      z = 0.f;
      const float* base = hin + (size_t)b * 64 * 192 + c;
      for (int i = 0; i < 64; ++i) z += aw[i] * base[i * 192];
    } else {
      z = gin[b * 4 + (c - 192)];
    }
    float s = g0[c] * rsqrtf(v0[c] + EPS_BN);
    pooled[b * 196 + c] = (z - m0[c]) * s + b0[c];
  }
}

// ---- dense head -----------------------------------------------------------
struct HeadArgs {
  const float *W1, *b1, *g1, *bb1, *m1, *v1;
  const float *W2, *b2, *g2, *bb2, *m2, *v2;
  const float *W3, *b3, *g3, *bb3, *m3, *v3;
  const float *Wo, *bo;
};

__global__ __launch_bounds__(128)
void head_kernel(const float* __restrict__ pooled, HeadArgs hw, float* __restrict__ out) {
  __shared__ float z[200], y1[384], y2[192], y3[96];
  int b = blockIdx.x, t = threadIdx.x;
  for (int c = t; c < 196; c += 128) z[c] = pooled[b * 196 + c];
  __syncthreads();
  for (int o = t; o < 384; o += 128) {
    float s = hw.b1[o];
    for (int k = 0; k < 196; ++k) s += z[k] * hw.W1[k * 384 + o];
    s = lrelu(s);
    float sc = hw.g1[o] * rsqrtf(hw.v1[o] + EPS_BN);
    y1[o] = (s - hw.m1[o]) * sc + hw.bb1[o];
  }
  __syncthreads();
  for (int o = t; o < 192; o += 128) {
    float s = hw.b2[o];
    for (int k = 0; k < 384; ++k) s += y1[k] * hw.W2[k * 192 + o];
    s = lrelu(s);
    float sc = hw.g2[o] * rsqrtf(hw.v2[o] + EPS_BN);
    y2[o] = (s - hw.m2[o]) * sc + hw.bb2[o];
  }
  __syncthreads();
  for (int o = t; o < 96; o += 128) {
    float s = hw.b3[o];
    for (int k = 0; k < 192; ++k) s += y2[k] * hw.W3[k * 96 + o];
    s = lrelu(s);
    float sc = hw.g3[o] * rsqrtf(hw.v3[o] + EPS_BN);
    y3[o] = (s - hw.m3[o]) * sc + hw.bb3[o];
  }
  __syncthreads();
  if (t < 4) {
    float s = hw.bo[t];
    for (int k = 0; k < 96; ++k) s += y3[k] * hw.Wo[k * 4 + t];
    out[b * 4 + t] = s;
  }
}

// ---- host -----------------------------------------------------------------
extern "C" void kernel_launch(void* const* d_in, const int* in_sizes, int n_in,
                              void* d_out, int out_size, void* d_ws, size_t ws_size,
                              hipStream_t stream) {
  (void)in_sizes; (void)n_in; (void)out_size; (void)ws_size;
  auto F32 = [&](int i) { return (const float*)d_in[i]; };

  // ---- workspace carve-up
  size_t offH = 0;
  auto allocH = [&](size_t nHalves) {
    _Float16* p = (_Float16*)d_ws + offH; offH += nHalves; return p;
  };
  _Float16* c1W2T = allocH(36864);
  _Float16* c1W3T = allocH(36864);
  _Float16 *W1dT[3], *W1bT[3], *W2T[3], *W3T[3], *WscT[3];
  for (int c = 0; c < 3; ++c) {
    W1dT[c] = allocH(36864); W1bT[c] = allocH(36864);
    W2T[c]  = allocH(36864); W3T[c]  = allocH(36864);
    WscT[c] = allocH(36864);
  }
  size_t ob = (offH * 2 + 255) & ~(size_t)255;
  float* h0 = (float*)((char*)d_ws + ob); ob += (size_t)32768 * 9 * 4;
  ob = (ob + 255) & ~(size_t)255;
  float* hA = (float*)((char*)d_ws + ob); ob += (size_t)32768 * 192 * 4;
  float* hB = (float*)((char*)d_ws + ob); ob += (size_t)32768 * 192 * 4;
  float* pooled = (float*)((char*)d_ws + ob); ob += (size_t)512 * 196 * 4;

  // ---- weight prep (fp16 transposed, folded W1a-W1b)
  const int PB = (36864 + 255) / 256;
  prep_wt<<<PB, 256, 0, stream>>>(F32(6 + 6),  nullptr, c1W2T, 192, 192);
  prep_wt<<<PB, 256, 0, stream>>>(F32(6 + 12), nullptr, c1W3T, 192, 192);
  for (int c = 0; c < 3; ++c) {
    int base = 6 + 24 * (c + 1);
    const float* W1 = F32(base + 0);  // (384,192)
    prep_wt<<<PB, 256, 0, stream>>>(W1, W1 + 192 * 192, W1dT[c], 192, 192);
    prep_wt<<<PB, 256, 0, stream>>>(W1 + 192 * 192, nullptr, W1bT[c], 192, 192);
    prep_wt<<<PB, 256, 0, stream>>>(F32(base + 6),  nullptr, W2T[c],  192, 192);
    prep_wt<<<PB, 256, 0, stream>>>(F32(base + 12), nullptr, W3T[c],  192, 192);
    prep_wt<<<PB, 256, 0, stream>>>(F32(base + 18), nullptr, WscT[c], 192, 192);
  }

  graphnorm_kernel<<<512, 64, 0, stream>>>(F32(0), F32(3), F32(4), F32(5), h0);

  auto mkargs = [&](int base, const _Float16* w1d, const _Float16* w1b,
                    const _Float16* w2t, const _Float16* w3t, const _Float16* wsct) {
    ConvArgs a;
    a.b1 = F32(base + 1);
    a.bn1g = F32(base + 2);  a.bn1b = F32(base + 3);  a.bn1m = F32(base + 4);  a.bn1v = F32(base + 5);
    a.b2 = F32(base + 7);
    a.bn2g = F32(base + 8);  a.bn2b = F32(base + 9);  a.bn2m = F32(base + 10); a.bn2v = F32(base + 11);
    a.b3 = F32(base + 13);
    a.bn3g = F32(base + 14); a.bn3b = F32(base + 15); a.bn3m = F32(base + 16); a.bn3v = F32(base + 17);
    a.bsc = F32(base + 19);
    a.scg = F32(base + 20);  a.scb = F32(base + 21);  a.scm = F32(base + 22);  a.scv = F32(base + 23);
    a.W1dT = w1d; a.W1bT = w1b; a.W2T = w2t; a.W3T = w3t; a.WscT = wsct;
    a.W1raw = F32(base + 0); a.Wscraw = F32(base + 18);
    return a;
  };

  conv_kernel<true ><<<512, 256, SMEM_FIRST, stream>>>(h0, hA, mkargs(6,  nullptr,  nullptr,  c1W2T,  c1W3T,  nullptr));
  conv_kernel<false><<<512, 256, SMEM_DYN,   stream>>>(hA, hB, mkargs(30, W1dT[0], W1bT[0], W2T[0], W3T[0], WscT[0]));
  conv_kernel<false><<<512, 256, SMEM_DYN,   stream>>>(hB, hA, mkargs(54, W1dT[1], W1bT[1], W2T[1], W3T[1], WscT[1]));
  conv_kernel<false><<<512, 256, SMEM_DYN,   stream>>>(hA, hB, mkargs(78, W1dT[2], W1bT[2], W2T[2], W3T[2], WscT[2]));

  attpool_kernel<<<512, 64, 0, stream>>>(hB, F32(2),
                                         F32(102), F32(103), F32(104), F32(105),
                                         F32(106), F32(107), F32(108), F32(109),
                                         pooled);

  HeadArgs hw;
  hw.W1 = F32(110); hw.b1 = F32(111); hw.g1 = F32(112); hw.bb1 = F32(113); hw.m1 = F32(114); hw.v1 = F32(115);
  hw.W2 = F32(116); hw.b2 = F32(117); hw.g2 = F32(118); hw.bb2 = F32(119); hw.m2 = F32(120); hw.v2 = F32(121);
  hw.W3 = F32(122); hw.b3 = F32(123); hw.g3 = F32(124); hw.bb3 = F32(125); hw.m3 = F32(126); hw.v3 = F32(127);
  hw.Wo = F32(128); hw.bo = F32(129);
  head_kernel<<<512, 128, 0, stream>>>(pooled, hw, (float*)d_out);
}
